// CTLSTM_90065464197525
// MI455X (gfx1250) — matrix operand order repaired and checked by reading, not compiled, not run on hardware
//
#include <hip/hip_runtime.h>
#include <stdint.h>

// dims (fixed by the reference)
#define B_ 64
#define L_ 512
#define I_ 256
#define H_ 256
#define G_ 1792   // 7*H
#define NT_ 112   // G/16 N-tiles

typedef __attribute__((ext_vector_type(16))) __bf16 v16bf;
typedef __attribute__((ext_vector_type(8)))  float  v8f;

__device__ __forceinline__ unsigned short f2bf(float f) {
    unsigned int u = __float_as_uint(f);
    u += 0x7FFFu + ((u >> 16) & 1u);          // round-to-nearest-even
    return (unsigned short)(u >> 16);
}
__device__ __forceinline__ float sigf(float v)  { return 1.0f / (1.0f + __expf(-v)); }
// tanh(x) = 1 - 2/(exp(2x)+1): exact at both saturation ends (exp->inf => 1, exp->0 => -1)
__device__ __forceinline__ float tanh_fast(float v) {
    return 1.0f - 2.0f / (__expf(2.0f * v) + 1.0f);
}
__device__ __forceinline__ float softplus_fast(float v) {
    return v > 15.0f ? v : __logf(1.0f + __expf(v));
}

// ---------------------------------------------------------------------------
// Kernel 0: f32 -> bf16 conversion of x, Wx, Wh into workspace
// ---------------------------------------------------------------------------
__global__ void conv_bf16_kernel(const float* __restrict__ x,
                                 const float* __restrict__ Wx,
                                 const float* __restrict__ Wh,
                                 unsigned short* __restrict__ xb,
                                 unsigned short* __restrict__ wxb,
                                 unsigned short* __restrict__ whb) {
    const long n_x = (long)B_ * L_ * I_;
    const long n_w = (long)G_ * I_;
    const long total = n_x + 2 * n_w;
    const long stride = (long)gridDim.x * blockDim.x;
    for (long i = (long)blockIdx.x * blockDim.x + threadIdx.x; i < total; i += stride) {
        if (i < n_x)              xb[i]              = f2bf(x[i]);
        else if (i < n_x + n_w)   wxb[i - n_x]       = f2bf(Wx[i - n_x]);
        else                      whb[i - n_x - n_w] = f2bf(Wh[i - n_x - n_w]);
    }
}

// ---------------------------------------------------------------------------
// Kernel 1: xg = x @ Wx^T  (bf16 WMMA, f32 accumulate), 4x4 register tiling.
// GEMM rows ordered m = l*64 + b; one wave-job = 64 rows (a full t) x 64 cols.
// Output stored in WMMA fragment order:
//   xg[(((t*4 + bt)*112 + ntile)*32 + lane)*8 + r]
// ---------------------------------------------------------------------------
__global__ __launch_bounds__(256) void xg_gemm_kernel(
        const unsigned short* __restrict__ xb,
        const unsigned short* __restrict__ wxb,
        float* __restrict__ xg) {
    const int lane = threadIdx.x & 31;
    const int warp = threadIdx.x >> 5;
    const int job  = blockIdx.x * 8 + warp;     // 512*28 = 14336 jobs
    const int mgrp = job / 28;                  // sequence position t = mgrp
    const int ngrp = job % 28;                  // 4 consecutive N tiles
    const int jl = lane & 15;
    const int ks = lane >> 4;

    const unsigned short* arow[4];
    #pragma unroll
    for (int mi = 0; mi < 4; ++mi)              // b = mi*16 + jl, l = mgrp
        arow[mi] = xb + ((size_t)(mi * 16 + jl) * L_ + mgrp) * I_;

    v8f zero = {0.f,0.f,0.f,0.f,0.f,0.f,0.f,0.f};
    v8f acc[4][4];
    #pragma unroll
    for (int mi = 0; mi < 4; ++mi)
        #pragma unroll
        for (int jj = 0; jj < 4; ++jj) acc[mi][jj] = zero;

    union frag { v16bf v; uint4 q[2]; };
    #pragma unroll
    for (int k = 0; k < 256; k += 32) {
        frag A[4], Bf[4];
        #pragma unroll
        for (int mi = 0; mi < 4; ++mi) {
            A[mi].q[0] = *(const uint4*)(arow[mi] + k + 8 * ks);
            A[mi].q[1] = *(const uint4*)(arow[mi] + k + 16 + 8 * ks);
        }
        #pragma unroll
        for (int jj = 0; jj < 4; ++jj) {
            int n = (ngrp * 4 + jj) * 16 + jl;
            const unsigned short* bp = wxb + (size_t)n * I_ + k + 8 * ks;
            Bf[jj].q[0] = *(const uint4*)bp;
            Bf[jj].q[1] = *(const uint4*)(bp + 16);
        }
        #pragma unroll
        for (int mi = 0; mi < 4; ++mi)
            #pragma unroll
            for (int jj = 0; jj < 4; ++jj)
                acc[mi][jj] = __builtin_amdgcn_wmma_f32_16x16x32_bf16(
                    false, A[mi].v, false, Bf[jj].v, (short)0, acc[mi][jj], false, false);
    }

    #pragma unroll
    for (int mi = 0; mi < 4; ++mi) {
        const size_t tb = ((size_t)mgrp * 4 + mi) * NT_;   // t = mgrp, bt = mi
        #pragma unroll
        for (int jj = 0; jj < 4; ++jj) {
            float* p = xg + ((tb + ngrp * 4 + jj) * 32 + lane) * 8;
            union { v8f v; float4 f[2]; } u;
            u.v = acc[mi][jj];
            *(float4*)p       = u.f[0];
            *((float4*)p + 1) = u.f[1];
        }
    }
}

// ---------------------------------------------------------------------------
// Kernel 2: sequential scan. grid = 4 (one WG per batch tile of 16),
// block = 256 (8 wave32). Wave w owns hidden slice j in [32w, 32w+32)
// across ALL 7 gates -> elementwise update is pure per-lane register math.
// h is exchanged through double-buffered LDS (bf16, A-matrix layout),
// one barrier per step. c lives in accumulator-layout registers.
// delta_t tile staged in LDS; next step's xg tile prefetched each step.
// ---------------------------------------------------------------------------
template <bool FUSED>
__global__ __launch_bounds__(256) void ctlstm_scan_kernel(
        const float* __restrict__ xg,              // fragment-ordered (pre path)
        const unsigned short* __restrict__ xb,     // bf16 x (fused path)
        const unsigned short* __restrict__ whb,    // bf16 Wh (G x I row-major)
        const unsigned short* __restrict__ wxb,    // bf16 Wx (fused path)
        const float* __restrict__ bx,
        const float* __restrict__ bh,
        const float* __restrict__ dt,
        const int*   __restrict__ slen,
        float* __restrict__ out) {
    __shared__ unsigned short hbuf[2][16 * 256];   // h state, bf16, double buffered (16KB)
    __shared__ unsigned short xlds[16 * 256];      // x_t staging, fused path (8KB)
    __shared__ float dtlds[16][512];               // delta_t tile (32KB)

    const int tid   = threadIdx.x;
    const int lane  = tid & 31;
    const int w     = tid >> 5;        // wave 0..7
    const int jlane = lane & 15;
    const int bsel  = lane >> 4;       // 0/1
    const int bt    = blockIdx.x;      // batch tile 0..3
    const size_t OS = (size_t)B_ * 513 * H_;   // per-output stride in d_out

    // bias (bx+bh) per (gate, 16-col block) -- scalar per lane
    float bias[7][2];
    #pragma unroll
    for (int g = 0; g < 7; ++g)
        #pragma unroll
        for (int hh = 0; hh < 2; ++hh) {
            int col = g * 256 + w * 32 + hh * 16 + jlane;
            bias[g][hh] = bx[col] + bh[col];
        }

    // seq lens per fragment row; zero the t=0 output slot (buffer is poisoned)
    int slr[8];
    #pragma unroll
    for (int r = 0; r < 8; ++r) {
        int bl = r + 8 * bsel;
        int bg = bt * 16 + bl;
        slr[r] = slen[bg];
        #pragma unroll
        for (int hh = 0; hh < 2; ++hh) {
            int j = w * 32 + hh * 16 + jlane;
            size_t ob = ((size_t)bg * 513) * H_ + j;
            #pragma unroll
            for (int k6 = 0; k6 < 6; ++k6) out[k6 * OS + ob] = 0.0f;
        }
    }

    // stage delta_t tile (coalesced) and zero initial h buffer
    for (int u = tid; u < 16 * 512; u += 256) {
        int bl = u >> 9, col = u & 511;
        dtlds[bl][col] = dt[(size_t)(bt * 16 + bl) * L_ + col];
    }
    for (int u = tid; u < 16 * 256; u += 256) hbuf[0][u] = 0;
    __syncthreads();

    v8f zero = {0.f,0.f,0.f,0.f,0.f,0.f,0.f,0.f};
    v8f c[2] = {zero, zero};           // cell state, accumulator layout

    for (int t = 0; t < L_; ++t) {
        const unsigned short* hsrc = hbuf[t & 1];
        unsigned short* hdst = hbuf[(t + 1) & 1];

        v8f acc[7][2];
        if constexpr (!FUSED) {
            // C-init = precomputed xg tile, fragment order: 2x b128 per tile
            const size_t tb = ((size_t)t * 4 + bt) * NT_;
            #pragma unroll
            for (int g = 0; g < 7; ++g)
                #pragma unroll
                for (int hh = 0; hh < 2; ++hh) {
                    int ntile = g * 16 + w * 2 + hh;
                    const float4* p = (const float4*)(xg + ((tb + ntile) * 32 + lane) * 8);
                    union { v8f v; float4 f[2]; } u;
                    u.f[0] = p[0];
                    u.f[1] = p[1];
                    acc[g][hh] = u.v;
                }
            // prefetch next step's xg tile (HBM-resident stream) -> global_prefetch
            if (t + 1 < L_) {
                const size_t tbn = ((size_t)(t + 1) * 4 + bt) * NT_;
                #pragma unroll
                for (int g = 0; g < 7; ++g)
                    #pragma unroll
                    for (int hh = 0; hh < 2; ++hh) {
                        int ntile = g * 16 + w * 2 + hh;
                        __builtin_prefetch(xg + ((tbn + ntile) * 32 + lane) * 8, 0, 1);
                    }
            }
        } else {
            #pragma unroll
            for (int g = 0; g < 7; ++g)
                #pragma unroll
                for (int hh = 0; hh < 2; ++hh) acc[g][hh] = zero;
            // stage x_t (16 x 256 bf16) into LDS, coalesced
            for (int uu = tid; uu < 16 * 256; uu += 256) {
                int bl = uu >> 8, col = uu & 255;
                xlds[uu] = xb[((size_t)(bt * 16 + bl) * L_ + t) * I_ + col];
            }
            __syncthreads();
        }

        // g += h(t-1) @ Wh^T : K = 256, 8 WMMA steps, 14 tiles per wave
        #pragma unroll
        for (int kk = 0; kk < 8; ++kk) {
            union { v16bf v; uint4 q[2]; } A;
            const unsigned short* ap = hsrc + jlane * 256 + kk * 32 + 8 * bsel;
            A.q[0] = *(const uint4*)ap;
            A.q[1] = *(const uint4*)(ap + 16);
            #pragma unroll
            for (int g = 0; g < 7; ++g)
                #pragma unroll
                for (int hh = 0; hh < 2; ++hh) {
                    int n = g * 256 + w * 32 + hh * 16 + jlane;
                    union { v16bf v; uint4 q[2]; } Bf;
                    const unsigned short* bp = whb + (size_t)n * I_ + kk * 32 + 8 * bsel;
                    Bf.q[0] = *(const uint4*)bp;
                    Bf.q[1] = *(const uint4*)(bp + 16);
                    acc[g][hh] = __builtin_amdgcn_wmma_f32_16x16x32_bf16(
                        false, A.v, false, Bf.v, (short)0, acc[g][hh], false, false);
                }
        }
        if constexpr (FUSED) {
            // g += x_t @ Wx^T
            #pragma unroll
            for (int kk = 0; kk < 8; ++kk) {
                union { v16bf v; uint4 q[2]; } A;
                const unsigned short* ap = xlds + jlane * 256 + kk * 32 + 8 * bsel;
                A.q[0] = *(const uint4*)ap;
                A.q[1] = *(const uint4*)(ap + 16);
                #pragma unroll
                for (int g = 0; g < 7; ++g)
                    #pragma unroll
                    for (int hh = 0; hh < 2; ++hh) {
                        int n = g * 256 + w * 32 + hh * 16 + jlane;
                        union { v16bf v; uint4 q[2]; } Bf;
                        const unsigned short* bp = wxb + (size_t)n * I_ + kk * 32 + 8 * bsel;
                        Bf.q[0] = *(const uint4*)bp;
                        Bf.q[1] = *(const uint4*)(bp + 16);
                        acc[g][hh] = __builtin_amdgcn_wmma_f32_16x16x32_bf16(
                            false, A.v, false, Bf.v, (short)0, acc[g][hh], false, false);
                    }
            }
        }

        // elementwise CT-LSTM update, per-lane registers only
        #pragma unroll
        for (int hh = 0; hh < 2; ++hh) {
            int j = w * 32 + hh * 16 + jlane;
            #pragma unroll
            for (int r = 0; r < 8; ++r) {
                int bl = r + 8 * bsel;
                int bg = bt * 16 + bl;
                float iv  = sigf(acc[0][hh][r] + bias[0][hh]);
                float fv  = sigf(acc[1][hh][r] + bias[1][hh]);
                float zv  = tanh_fast(acc[2][hh][r] + bias[2][hh]);
                float ov  = sigf(acc[3][hh][r] + bias[3][hh]);
                float dv  = softplus_fast(acc[4][hh][r] + bias[4][hh]);
                float ibv = sigf(acc[5][hh][r] + bias[5][hh]);
                float fbv = sigf(acc[6][hh][r] + bias[6][hh]);
                float cold  = c[hh][r];
                float cnew  = fv * cold + iv * zv;
                float hdisc = ov * tanh_fast(cnew);
                float cbar  = fbv * cold + ibv * zv;
                float dtv   = dtlds[bl][t];
                float ct    = cbar + (cnew - cbar) * __expf(-dv * dtv);
                float ht    = ov * tanh_fast(ct);
                bool  msk   = t < slr[r];
                float hn = msk ? ht : 0.0f;
                float cn = msk ? ct : 0.0f;
                size_t ob = ((size_t)bg * 513 + (t + 1)) * H_ + j;
                out[0 * OS + ob] = hn;
                out[1 * OS + ob] = msk ? hdisc : 0.0f;
                out[2 * OS + ob] = msk ? cnew  : 0.0f;
                out[3 * OS + ob] = msk ? cbar  : 0.0f;
                out[4 * OS + ob] = msk ? ov    : 0.0f;
                out[5 * OS + ob] = msk ? dv    : 0.0f;
                hdst[bl * 256 + j] = f2bf(hn);
                c[hh][r] = cn;
            }
        }
        __syncthreads();   // h(t) visible to all waves; all reads of hsrc done
    }
}

// ---------------------------------------------------------------------------
// Host launch
// ---------------------------------------------------------------------------
extern "C" void kernel_launch(void* const* d_in, const int* in_sizes, int n_in,
                              void* d_out, int out_size, void* d_ws, size_t ws_size,
                              hipStream_t stream) {
    const float* x    = (const float*)d_in[0];
    const float* dt   = (const float*)d_in[1];
    const int*   slen = (const int*)d_in[2];
    const float* Wx   = (const float*)d_in[3];
    const float* bx   = (const float*)d_in[4];
    const float* Wh   = (const float*)d_in[5];
    const float* bh   = (const float*)d_in[6];
    float* out = (float*)d_out;

    const size_t XG_BYTES = (size_t)L_ * B_ * G_ * sizeof(float);  // 235 MB
    const size_t XB_BYTES = (size_t)B_ * L_ * I_ * 2;              // 16 MB
    const size_t W_BYTES  = (size_t)G_ * I_ * 2;                   // 0.9 MB
    const bool pre = ws_size >= XG_BYTES + XB_BYTES + 2 * W_BYTES;

    char* ws = (char*)d_ws;
    float* xg = (float*)ws;
    unsigned short* xb  = (unsigned short*)(pre ? (ws + XG_BYTES) : ws);
    unsigned short* wxb = (unsigned short*)((char*)xb + XB_BYTES);
    unsigned short* whb = (unsigned short*)((char*)wxb + W_BYTES);

    conv_bf16_kernel<<<2048, 256, 0, stream>>>(x, Wx, Wh, xb, wxb, whb);
    if (pre) {
        // 512 M-groups (64 rows) x 28 N-groups = 14336 wave jobs / 8 = 1792 blocks
        xg_gemm_kernel<<<1792, 256, 0, stream>>>(xb, wxb, xg);
        ctlstm_scan_kernel<false><<<4, 256, 0, stream>>>(xg, xb, whb, wxb,
                                                         bx, bh, dt, slen, out);
    } else {
        ctlstm_scan_kernel<true><<<4, 256, 0, stream>>>(nullptr, xb, whb, wxb,
                                                        bx, bh, dt, slen, out);
    }
}
